// UDCUNet_62294205661416
// MI455X (gfx1250) — matrix operand order repaired and checked by reading, not compiled, not run on hardware
//
#include <hip/hip_runtime.h>

typedef _Float16 half_t;
typedef __attribute__((ext_vector_type(16))) _Float16 v16h;
typedef __attribute__((ext_vector_type(8)))  float    v8f;

union F16Frag { uint4 u[2]; v16h v; };

// ---- CDNA5 async global->LDS copy (ASYNCcnt-tracked), guarded by probe ----
#if defined(__has_builtin)
#if __has_builtin(__builtin_amdgcn_global_load_async_to_lds_b128) && \
    __has_builtin(__builtin_amdgcn_s_wait_asynccnt)
#define UDC_ASYNC 1
#endif
#endif

#if defined(UDC_ASYNC)
typedef int v4i_t __attribute__((vector_size(16)));
typedef v4i_t __attribute__((address_space(1)))* gas1_p;   // global int4*
typedef v4i_t __attribute__((address_space(3)))* las3_p;   // LDS int4*
__device__ __forceinline__ void async_cp16(const half_t* g, half_t* l) {
  // generic->AS1: numeric identity for global; generic->AS3: low 32 bits are the LDS offset
  __builtin_amdgcn_global_load_async_to_lds_b128(
      (gas1_p)(uintptr_t)g, (las3_p)(uint32_t)(uintptr_t)l, 0, 0);
}
#endif

struct ConvArgs {
  const half_t* in;      // HWC f16, cinp channels
  const half_t* wpk;     // [cout][taps][cinp] f16
  const float*  bias;    // [cout] fp32
  const half_t* scale;   // optional SFT scale (HWC, cinp ch)
  const half_t* shift;   // optional SFT shift
  const half_t* add;     // optional residual (HWC, cout ch, output res)
  const float*  xres;    // final-layer: original x, NCHW fp32
  float*        out32;   // final-layer output (NCHW fp32)
  half_t*       out;     // normal output (HWC f16)
  int W, H, Wo, Ho, cinp, cout, ksz, stride, act, outmode;
};

// worst cases: k3 s1 c256 -> 3*34*256 ; k3 s2 c128 -> 3*66*128 ; both <= 26112 (52 KB)
#define SMEM_ELEMS 26112

__device__ __forceinline__ float actf(float v, int act) {
  if (act == 1) return fmaxf(v, 0.f);
  if (act == 2) return (v >= 0.f) ? v : 0.1f * v;
  if (act == 3) return (v >= 0.f) ? v : 0.2f * v;
  return v;
}

// ---------------------------------------------------------------------------
// Implicit-GEMM conv, WMMA f32 <- f16 x f16 (16x16x32).
// Block = 128 threads = 4 wave32 waves; block tile = 64 Cout x 32 pixels.
// Each wave: 16 Cout x 32 pixels as TWO accumulators sharing one A fragment
// (2x weight-stream reuse). Activation patch staged in LDS (HWC, tap-major K)
// via GLOBAL_LOAD_ASYNC_TO_LDS_B128 when no SFT fusion is needed.
// ---------------------------------------------------------------------------
__global__ __launch_bounds__(128) void k_conv(ConvArgs A) {
  __shared__ half_t smem[SMEM_ELEMS];

  const int ks     = A.ksz;
  const int halo   = (ks == 3) ? 1 : 0;
  const int rows   = (ks == 3) ? 3 : 1;
  const int tw     = 32 * A.stride + 2 * halo;
  const int tilesX = A.Wo >> 5;                 // 32-pixel tiles
  const int ty     = blockIdx.x / tilesX;       // output row
  const int x0     = (blockIdx.x - ty * tilesX) << 5;
  const int tid    = threadIdx.x;
  const int cinp   = A.cinp;

  const int total = rows * tw * cinp;
#if defined(UDC_ASYNC)
  if (A.scale == nullptr) {
    // ---- async DMA staging: memory -> LDS, no VGPR round trip ----
    for (int e = tid * 8; e < total; e += 128 * 8) {
      const int c   = e % cinp;
      const int pos = e / cinp;
      const int xx  = pos % tw;
      const int yy  = pos / tw;
      const int yi  = ty * A.stride + yy - halo;
      const int xi  = x0 * A.stride + xx - halo;
      if (yi >= 0 && yi < A.H && xi >= 0 && xi < A.W) {
        const size_t off = ((size_t)yi * A.W + xi) * (size_t)cinp + c;
        async_cp16(A.in + off, smem + e);
      } else {
        const uint4 z = {0u, 0u, 0u, 0u};
        *(uint4*)(smem + e) = z;                // zero halo (DScnt, covered by barrier)
      }
    }
    __builtin_amdgcn_s_wait_asynccnt(0);
  } else
#endif
  {
    // ---- VALU staging path, fusing SFT: fea*(1+scale)+shift ----
    for (int e = tid * 8; e < total; e += 128 * 8) {
      const int c   = e % cinp;
      const int pos = e / cinp;
      const int xx  = pos % tw;
      const int yy  = pos / tw;
      const int yi  = ty * A.stride + yy - halo;
      const int xi  = x0 * A.stride + xx - halo;
      uint4 val = {0u, 0u, 0u, 0u};
      if (yi >= 0 && yi < A.H && xi >= 0 && xi < A.W) {
        const size_t off = ((size_t)yi * A.W + xi) * (size_t)cinp + c;
        val = *(const uint4*)(A.in + off);
        if (A.scale) {
          const uint4 sc = *(const uint4*)(A.scale + off);
          const uint4 sh = *(const uint4*)(A.shift + off);
          half_t*       v = (half_t*)&val;
          const half_t* s = (const half_t*)&sc;
          const half_t* h = (const half_t*)&sh;
#pragma unroll
          for (int q = 0; q < 8; ++q)
            v[q] = (half_t)((float)v[q] * (1.0f + (float)s[q]) + (float)h[q]);
        }
      }
      *(uint4*)(smem + e) = val;
    }
  }
  __syncthreads();

  // ---- WMMA main loop ----
  const int lane   = tid & 31;
  const int wave   = tid >> 5;
  const int hi     = lane >> 4;     // half-wave selector
  const int mn     = lane & 15;     // A: row (Cout), B: col (pixel)
  const int coBase = blockIdx.y * 64 + wave * 16;
  if (coBase >= A.cout) return;     // no barriers beyond this point

  const int taps = ks * ks;
  v8f acc0 = {0.f,0.f,0.f,0.f,0.f,0.f,0.f,0.f};
  v8f acc1 = {0.f,0.f,0.f,0.f,0.f,0.f,0.f,0.f};

  int coc = coBase + mn;
  if (coc >= A.cout) coc = A.cout - 1;            // clamp (stores are guarded)
  const half_t* wbase = A.wpk + (size_t)coc * taps * cinp;
  const int sub1 = 16 * A.stride * cinp;          // LDS offset of 2nd pixel subtile

  for (int tap = 0; tap < taps; ++tap) {
    const int dy = tap / ks;
    const int dx = tap - dy * ks;
    const half_t* wr = wbase + (size_t)tap * cinp + hi * 16;
    const half_t* br = smem + (size_t)(dy * tw + mn * A.stride + dx) * cinp + hi * 16;
    if (tap + 1 < taps) __builtin_prefetch(wr + cinp, 0, 1);  // next tap's weights
    for (int c0 = 0; c0 < cinp; c0 += 32) {
      F16Frag a, b0, b1;
      // A (16x32 f16): row = lane&15, 16 contiguous K starting at hi*16
      a.u[0] = *(const uint4*)(wr + c0);
      a.u[1] = *(const uint4*)(wr + c0 + 8);
      // B (32x16 f16): col = lane&15, 16 contiguous K starting at hi*16
      b0.u[0] = *(const uint4*)(br + c0);
      b0.u[1] = *(const uint4*)(br + c0 + 8);
      b1.u[0] = *(const uint4*)(br + sub1 + c0);
      b1.u[1] = *(const uint4*)(br + sub1 + c0 + 8);
      acc0 = __builtin_amdgcn_wmma_f32_16x16x32_f16(
          false, a.v, false, b0.v, (short)0, acc0, false, false);
      acc1 = __builtin_amdgcn_wmma_f32_16x16x32_f16(
          false, a.v, false, b1.v, (short)0, acc1, false, false);
    }
  }

  // ---- epilogue: bias + residual + activation ----
  const int p0 = ty * A.Wo + x0 + mn;
  v8f accs[2] = {acc0, acc1};
  if (A.outmode == 0 && coBase + 16 <= A.cout) {
    // fast path: fully vectorized b128 epilogue
    const int cb = coBase + hi * 8;
    float bs[8];
    *(float4*)&bs[0] = *(const float4*)(A.bias + cb);
    *(float4*)&bs[4] = *(const float4*)(A.bias + cb + 4);
#pragma unroll
    for (int t = 0; t < 2; ++t) {
      const int p = p0 + t * 16;
      const v8f acc = accs[t];
      half_t o[8];
      if (A.add) {
        const uint4 av = *(const uint4*)(A.add + (size_t)p * A.cout + cb);
        const half_t* ah = (const half_t*)&av;
#pragma unroll
        for (int i = 0; i < 8; ++i)
          o[i] = (half_t)actf(acc[i] + bs[i] + (float)ah[i], A.act);
      } else {
#pragma unroll
        for (int i = 0; i < 8; ++i)
          o[i] = (half_t)actf(acc[i] + bs[i], A.act);
      }
      *(uint4*)(A.out + (size_t)p * A.cout + cb) = *(uint4*)o;
    }
  } else {
    // guarded scalar path (partial Cout tile / final fp32 NCHW layer)
#pragma unroll
    for (int t = 0; t < 2; ++t) {
      const int p = p0 + t * 16;
      const v8f acc = accs[t];
#pragma unroll
      for (int i = 0; i < 8; ++i) {
        const int co = coBase + i + hi * 8;
        if (co >= A.cout) continue;
        float v = acc[i] + A.bias[co];
        if (A.add) v += (float)A.add[(size_t)p * A.cout + co];
        v = actf(v, A.act);
        if (A.outmode == 0) {
          A.out[(size_t)p * A.cout + co] = (half_t)v;
        } else {
          const size_t q = (size_t)co * A.Ho * A.Wo + p;
          A.out32[q] = fmaxf(A.xres[q] + v, 0.f);     // relu(x + conv_last)
        }
      }
    }
  }
}

// ---------------------------------------------------------------------------
// Weight repack: OIHW fp32 -> [co][tap][ci_pad] f16 (zero-padded channels)
// ---------------------------------------------------------------------------
__global__ void k_repack(const float* __restrict__ w, half_t* __restrict__ wpk,
                         int cout, int cin, int cinp, int taps) {
  const int i = blockIdx.x * blockDim.x + threadIdx.x;
  const int total = cout * taps * cinp;
  if (i >= total) return;
  const int ci  = i % cinp;
  const int r   = i / cinp;
  const int tap = r % taps;
  const int co  = r / taps;
  float v = (ci < cin) ? w[((size_t)co * cin + ci) * taps + tap] : 0.f;
  wpk[i] = (half_t)v;
}

// ---------------------------------------------------------------------------
// Build f16 HWC (32-ch padded) versions of x and concat([x, psf])
// ---------------------------------------------------------------------------
__global__ void k_prep(const float* __restrict__ x, half_t* __restrict__ x32,
                       half_t* __restrict__ xpsf, int HW) {
  const int p = blockIdx.x * blockDim.x + threadIdx.x;
  if (p >= HW) return;
  const float psf[5] = {-0.002f, 0.0352f, -2.0215e-05f, 0.006f, 9.4328e-05f};
#pragma unroll
  for (int c = 0; c < 32; ++c) {
    float v = (c < 3) ? x[(size_t)c * HW + p] : 0.f;
    x32[(size_t)p * 32 + c] = (half_t)v;
    float v2 = (c < 3) ? v : ((c < 8) ? psf[c - 3] : 0.f);
    xpsf[(size_t)p * 32 + c] = (half_t)v2;
  }
}

// ---------------------------------------------------------------------------
// Per-pixel dynamic 3x3 (FAC) conv, edge-clamped; 8 channels per thread so all
// feat and dynamic-kernel traffic is b128. Fused add: plain (baseMode 0) or
// pixel-shuffle r=2 from a 4C low-res buffer (baseMode 1).
// ker channel layout: c*9 + i*3 + j  -> 72 contiguous halves per 8-ch chunk.
// ---------------------------------------------------------------------------
__global__ void k_dynconv(const half_t* __restrict__ feat, const half_t* __restrict__ ker,
                          const half_t* __restrict__ base, int baseMode,
                          half_t* __restrict__ out, int H, int W, int C) {
  const int i = blockIdx.x * blockDim.x + threadIdx.x;
  const int chunks = C >> 3;
  const int total = H * W * chunks;
  if (i >= total) return;
  const int ch = (i % chunks) << 3;
  const int p  = i / chunks;
  const int x  = p % W;
  const int y  = p / W;

  float acc[8];
  if (baseMode == 0) {
    const uint4 bv = *(const uint4*)(base + (size_t)p * C + ch);
    const half_t* bh = (const half_t*)&bv;
#pragma unroll
    for (int q = 0; q < 8; ++q) acc[q] = (float)bh[q];
  } else {
    const int Wl = W >> 1;
    const size_t pl = (size_t)(y >> 1) * Wl + (x >> 1);
    const size_t bb = pl * (size_t)(4 * C) + (size_t)ch * 4 + (y & 1) * 2 + (x & 1);
#pragma unroll
    for (int q = 0; q < 8; ++q) acc[q] = (float)base[bb + q * 4];
  }

  // 72 contiguous dynamic-kernel taps for this 8-channel chunk
  uint4 kv[9];
  const half_t* kp = ker + (size_t)p * (C * 9) + (size_t)ch * 9;
#pragma unroll
  for (int t = 0; t < 9; ++t) kv[t] = *(const uint4*)(kp + t * 8);
  const half_t* kk = (const half_t*)kv;

#pragma unroll
  for (int t = 0; t < 9; ++t) {
    int yy = y + t / 3 - 1; yy = (yy < 0) ? 0 : ((yy > H - 1) ? H - 1 : yy);
    int xx = x + t % 3 - 1; xx = (xx < 0) ? 0 : ((xx > W - 1) ? W - 1 : xx);
    const uint4 fv = *(const uint4*)(feat + ((size_t)yy * W + xx) * C + ch);
    const half_t* fh = (const half_t*)&fv;
#pragma unroll
    for (int q = 0; q < 8; ++q) acc[q] += (float)fh[q] * (float)kk[q * 9 + t];
  }

  half_t o[8];
#pragma unroll
  for (int q = 0; q < 8; ++q) o[q] = (half_t)acc[q];
  *(uint4*)(out + (size_t)p * C + ch) = *(uint4*)o;
}

// ---------------------------------------------------------------------------
// Host orchestration
// ---------------------------------------------------------------------------
extern "C" void kernel_launch(void* const* d_in, const int* in_sizes, int n_in,
                              void* d_out, int out_size, void* d_ws, size_t ws_size,
                              hipStream_t stream) {
  (void)in_sizes; (void)n_in; (void)out_size; (void)ws_size;
  const float* x = (const float*)d_in[0];

  struct CP   { const float* b; const float* w; };
  struct SFT4 { CP s0, s1, t0, t1; };
  struct Blk  { CP conv0, conv1; SFT4 sft0, sft1; };
  struct Res  { CP c1, c2; };

  int idx = 1;  // d_in[0] = x; params follow in JAX sorted-pytree order
  auto rd  = [&]() { CP p; p.b = (const float*)d_in[idx++]; p.w = (const float*)d_in[idx++]; return p; };
  auto rdS = [&]() { SFT4 s; s.s0 = rd(); s.s1 = rd(); s.t0 = rd(); s.t1 = rd(); return s; };
  auto rdB = [&]() { Blk b; b.conv0 = rd(); b.conv1 = rd(); b.sft0 = rdS(); b.sft1 = rdS(); return b; };
  auto rdR = [&]() { Res r; r.c1 = rd(); r.c2 = rd(); return r; };

  // sorted keys: CondNet0-3, KNet0-3, bottom, cond_first, cond_head, conv_first,
  // conv_last, dec0-2, down0-2, enc0-2, k_first, k_head, up0-2
  CP Cond[4][4], KN[4][4];
  for (int g = 0; g < 4; ++g) for (int i = 0; i < 4; ++i) Cond[g][i] = rd();
  for (int g = 0; g < 4; ++g) for (int i = 0; i < 4; ++i) KN[g][i]   = rd();
  Blk bottom[8]; for (int i = 0; i < 8; ++i) bottom[i] = rdB();
  Res cfirst[2]; cfirst[0] = rdR(); cfirst[1] = rdR();
  CP cond_head  = rd();
  CP conv_first = rd();
  CP conv_last  = rd();
  Blk dec0[2]; dec0[0] = rdB(); dec0[1] = rdB();
  Blk dec1[2]; dec1[0] = rdB(); dec1[1] = rdB();
  Blk dec2[2]; dec2[0] = rdB(); dec2[1] = rdB();
  CP down0 = rd(), down1 = rd(), down2 = rd();
  Blk enc0[2]; enc0[0] = rdB(); enc0[1] = rdB();
  Blk enc1[2]; enc1[0] = rdB(); enc1[1] = rdB();
  Blk enc2[2]; enc2[0] = rdB(); enc2[1] = rdB();
  Res kfirstP[2]; kfirstP[0] = rdR(); kfirstP[1] = rdR();
  CP k_headP = rd();
  CP up0 = rd(), up1 = rd(), up2 = rd();

  // deterministic bump allocator over d_ws
  char* wsp = (char*)d_ws;
  size_t woff = 0;
  auto alloc = [&](size_t bytes) -> void* {
    void* p = wsp + woff;
    woff += (bytes + 255) & ~(size_t)255;
    return p;
  };

  auto convL = [&](CP p, const half_t* in, int Wd, int Hd, int cin, int cout, int k, int s,
                   int act, const half_t* scale, const half_t* shift, const half_t* add,
                   float* out32, const float* xres) -> half_t* {
    const int cinp = (cin + 31) & ~31;
    const int taps = k * k;
    half_t* wpk = (half_t*)alloc((size_t)cout * taps * cinp * sizeof(half_t));
    const int tot = cout * taps * cinp;
    k_repack<<<(tot + 255) / 256, 256, 0, stream>>>(p.w, wpk, cout, cin, cinp, taps);
    const int Wo = Wd / s, Ho = Hd / s;
    half_t* out = nullptr;
    if (!out32) out = (half_t*)alloc((size_t)Wo * Ho * cout * sizeof(half_t));
    ConvArgs A;
    A.in = in; A.wpk = wpk; A.bias = p.b; A.scale = scale; A.shift = shift; A.add = add;
    A.xres = xres; A.out32 = out32; A.out = out;
    A.W = Wd; A.H = Hd; A.Wo = Wo; A.Ho = Ho; A.cinp = cinp; A.cout = cout;
    A.ksz = k; A.stride = s; A.act = act; A.outmode = out32 ? 1 : 0;
    dim3 g((Wo / 32) * Ho, (cout + 63) / 64);
    k_conv<<<g, 128, 0, stream>>>(A);
    return out;
  };

  auto resblk = [&](const Res& r, half_t* xb, int C, int Wd, int Hd) -> half_t* {
    half_t* t = convL(r.c1, xb, Wd, Hd, C, C, 3, 1, 1, 0, 0, 0, 0, 0);       // relu(conv)
    return     convL(r.c2, t,  Wd, Hd, C, C, 3, 1, 0, 0, 0, xb, 0, 0);       // + x
  };

  auto sftblk = [&](const Blk& b, half_t* fea, half_t* cond, int C, int Wd, int Hd) -> half_t* {
    half_t* t0  = convL(b.sft0.s0, cond, Wd, Hd, C, C, 1, 1, 2, 0, 0, 0, 0, 0); // lrelu 0.1
    half_t* sc0 = convL(b.sft0.s1, t0,   Wd, Hd, C, C, 1, 1, 0, 0, 0, 0, 0, 0);
    half_t* t1  = convL(b.sft0.t0, cond, Wd, Hd, C, C, 1, 1, 2, 0, 0, 0, 0, 0);
    half_t* sh0 = convL(b.sft0.t1, t1,   Wd, Hd, C, C, 1, 1, 0, 0, 0, 0, 0, 0);
    half_t* f   = convL(b.conv0, fea, Wd, Hd, C, C, 3, 1, 1, sc0, sh0, 0, 0, 0); // SFT->conv->relu
    half_t* t2  = convL(b.sft1.s0, cond, Wd, Hd, C, C, 1, 1, 2, 0, 0, 0, 0, 0);
    half_t* sc1 = convL(b.sft1.s1, t2,   Wd, Hd, C, C, 1, 1, 0, 0, 0, 0, 0, 0);
    half_t* t3  = convL(b.sft1.t0, cond, Wd, Hd, C, C, 1, 1, 2, 0, 0, 0, 0, 0);
    half_t* sh1 = convL(b.sft1.t1, t3,   Wd, Hd, C, C, 1, 1, 0, 0, 0, 0, 0, 0);
    return convL(b.conv1, f, Wd, Hd, C, C, 3, 1, 0, sc1, sh1, fea, 0, 0);        // SFT->conv + fea
  };

  auto seq4 = [&](CP* net, half_t* in, int Wd, int Hd,
                  const int* ksv, const int* sv, const int* coutv) -> half_t* {
    half_t* cur = in; int W2 = Wd, H2 = Hd, cin = 32;
    for (int i = 0; i < 4; ++i) {
      cur = convL(net[i], cur, W2, H2, cin, coutv[i], ksv[i], sv[i],
                  (i < 3) ? 3 : 0, 0, 0, 0, 0, 0);  // lrelu 0.2 between convs
      W2 /= sv[i]; H2 /= sv[i]; cin = coutv[i];
    }
    return cur;
  };

  auto dynconv = [&](half_t* feat, half_t* ker, half_t* base, int baseMode,
                     int Hd, int Wd, int C) -> half_t* {
    half_t* o = (half_t*)alloc((size_t)Hd * Wd * C * sizeof(half_t));
    const int tot = Hd * Wd * (C >> 3);
    k_dynconv<<<(tot + 255) / 256, 256, 0, stream>>>(feat, ker, base, baseMode, o, Hd, Wd, C);
    return o;
  };

  // ---- inputs ----
  const int HW = 512 * 512;
  half_t* x32  = (half_t*)alloc((size_t)HW * 32 * sizeof(half_t));
  half_t* xpsf = (half_t*)alloc((size_t)HW * 32 * sizeof(half_t));
  k_prep<<<(HW + 255) / 256, 256, 0, stream>>>(x, x32, xpsf, HW);

  static const int S0s[4] = {1,1,1,1}, S1s[4] = {2,1,1,1}, S2s[4] = {2,2,1,1}, S3s[4] = {2,2,2,1};
  static const int K0k[4] = {1,1,1,1}, K1k[4] = {3,1,1,1}, K2k[4] = {3,3,1,1}, K3k[4] = {3,3,3,1};
  const int KC0[4] = {32,32,32,288},  KC1[4] = {32,32,32,576};
  const int KC2[4] = {32,32,32,1152}, KC3[4] = {32,32,32,2304};
  const int CC0[4] = {32,32,32,32},   CC1[4] = {32,32,32,64};
  const int CC2[4] = {32,32,32,128},  CC3[4] = {32,32,32,256};

  // ---- dynamic-kernel branch ----
  half_t* kf = convL(k_headP, xpsf, 512, 512, 8, 32, 3, 1, 3, 0, 0, 0, 0, 0);
  kf = resblk(kfirstP[0], kf, 32, 512, 512);
  kf = resblk(kfirstP[1], kf, 32, 512, 512);
  half_t* kernel0 = seq4(KN[0], kf, 512, 512, K0k, S0s, KC0);
  half_t* kernel1 = seq4(KN[1], kf, 512, 512, K1k, S1s, KC1);
  half_t* kernel2 = seq4(KN[2], kf, 512, 512, K2k, S2s, KC2);
  half_t* kernel3 = seq4(KN[3], kf, 512, 512, K3k, S3s, KC3);

  // ---- condition branch ----
  half_t* cnd = convL(cond_head, x32, 512, 512, 3, 32, 3, 1, 3, 0, 0, 0, 0, 0);
  cnd = resblk(cfirst[0], cnd, 32, 512, 512);
  cnd = resblk(cfirst[1], cnd, 32, 512, 512);
  half_t* cond0 = seq4(Cond[0], cnd, 512, 512, K0k, S0s, CC0);
  half_t* cond1 = seq4(Cond[1], cnd, 512, 512, K1k, S1s, CC1);
  half_t* cond2 = seq4(Cond[2], cnd, 512, 512, K2k, S2s, CC2);
  half_t* cond3 = seq4(Cond[3], cnd, 512, 512, K3k, S3s, CC3);

  // ---- encoder ----
  half_t* fea0 = convL(conv_first, x32, 512, 512, 3, 32, 3, 1, 3, 0, 0, 0, 0, 0);
  fea0 = sftblk(enc0[0], fea0, cond0, 32, 512, 512);
  fea0 = sftblk(enc0[1], fea0, cond0, 32, 512, 512);
  half_t* d0 = convL(down0, fea0, 512, 512, 32, 64, 3, 2, 0, 0, 0, 0, 0, 0);
  half_t* fea1 = d0;
  fea1 = sftblk(enc1[0], fea1, cond1, 64, 256, 256);
  fea1 = sftblk(enc1[1], fea1, cond1, 64, 256, 256);
  half_t* d1 = convL(down1, fea1, 256, 256, 64, 128, 3, 2, 0, 0, 0, 0, 0, 0);
  half_t* fea2 = d1;
  fea2 = sftblk(enc2[0], fea2, cond2, 128, 128, 128);
  fea2 = sftblk(enc2[1], fea2, cond2, 128, 128, 128);
  half_t* d2 = convL(down2, fea2, 128, 128, 128, 256, 3, 2, 0, 0, 0, 0, 0, 0);

  // ---- bottom ----
  half_t* fb = d2;
  for (int i = 0; i < 8; ++i) fb = sftblk(bottom[i], fb, cond3, 256, 64, 64);
  half_t* fb2 = dynconv(d2, kernel3, fb, 0, 64, 64, 256);

  // ---- decoder ----
  half_t* u2c = convL(up2, fb2, 64, 64, 256, 512, 3, 1, 0, 0, 0, 0, 0, 0);
  half_t* de2 = dynconv(fea2, kernel2, u2c, 1, 128, 128, 128);
  de2 = sftblk(dec2[0], de2, cond2, 128, 128, 128);
  de2 = sftblk(dec2[1], de2, cond2, 128, 128, 128);
  half_t* u1c = convL(up1, de2, 128, 128, 128, 256, 3, 1, 0, 0, 0, 0, 0, 0);
  half_t* de1 = dynconv(fea1, kernel1, u1c, 1, 256, 256, 64);
  de1 = sftblk(dec1[0], de1, cond1, 64, 256, 256);
  de1 = sftblk(dec1[1], de1, cond1, 64, 256, 256);
  half_t* u0c = convL(up0, de1, 256, 256, 64, 128, 3, 1, 0, 0, 0, 0, 0, 0);
  half_t* de0 = dynconv(fea0, kernel0, u0c, 1, 512, 512, 32);
  de0 = sftblk(dec0[0], de0, cond0, 32, 512, 512);
  de0 = sftblk(dec0[1], de0, cond0, 32, 512, 512);

  // ---- final: relu(x + conv_last(de0)) -> fp32 NCHW d_out ----
  convL(conv_last, de0, 512, 512, 32, 3, 3, 1, 0, 0, 0, 0, (float*)d_out, x);
}
